// SmolVLMWithExpert_2619930051345
// MI455X (gfx1250) — matrix operand order repaired and checked by reading, not compiled, not run on hardware
//
#include <hip/hip_runtime.h>
#include <math.h>

typedef __attribute__((ext_vector_type(16))) _Float16    v16h;
typedef __attribute__((ext_vector_type(8)))  float        v8f;
typedef __attribute__((ext_vector_type(8)))  unsigned int v8u;

union FragU { v8u u; v16h h; };

#define NUM_HEADS 15
#define NUM_KV    5
#define GROUPS    3
#define HDIM      64
#define LSEQ      2048

// LDS strides (padded to avoid bank conflicts)
#define KSTR_H  72   // halves per key row of roped-K (64 data + 8 pad)
#define KSTR_DW 36
#define VSTR_H  40   // halves per d-row of V^T (32 keys + 8 pad)
#define VSTR_DW 20
#define PSTR_H  40   // halves per M-row of P (32 keys + 8 pad)
#define PSTR_DW 20

#define LN1E4_OVER_32 0.2878231366242557f   // ln(10000)/32
#define NEG_BIG (-3.402823466e38f)

__device__ __forceinline__ unsigned short f2h(float f) {
  _Float16 h = (_Float16)f;
  return __builtin_bit_cast(unsigned short, h);
}

// max over the 16-lane row group (lanes 0-15 and 16-31 reduce independently),
// using DPP butterflies (VALU) instead of ds_bpermute (LDS pipe).
__device__ __forceinline__ float row16_max(float x) {
  int t;
  t = __builtin_amdgcn_update_dpp(0, __float_as_int(x), 0xB1, 0xF, 0xF, true);  // quad_perm [1,0,3,2]
  x = fmaxf(x, __int_as_float(t));
  t = __builtin_amdgcn_update_dpp(0, __float_as_int(x), 0x4E, 0xF, 0xF, true);  // quad_perm [2,3,0,1]
  x = fmaxf(x, __int_as_float(t));
  t = __builtin_amdgcn_update_dpp(0, __float_as_int(x), 0x141, 0xF, 0xF, true); // row_half_mirror
  x = fmaxf(x, __int_as_float(t));
  t = __builtin_amdgcn_update_dpp(0, __float_as_int(x), 0x140, 0xF, 0xF, true); // row_mirror
  x = fmaxf(x, __int_as_float(t));
  return x;
}

__launch_bounds__(128)
__global__ void fa_rope_wmma_kernel(const float* __restrict__ Q,
                                    const float* __restrict__ K,
                                    const float* __restrict__ V,
                                    const int*   __restrict__ pos,
                                    float* __restrict__ out) {
  __shared__ unsigned int KldsU[32 * KSTR_DW];       // roped K, f16
  __shared__ unsigned int VtU[HDIM * VSTR_DW];       // V transposed, f16
  __shared__ unsigned int PldsU[4 * 16 * PSTR_DW];   // per-wave P staging, f16
  unsigned short* Klds16 = reinterpret_cast<unsigned short*>(KldsU);
  unsigned short* Vt16   = reinterpret_cast<unsigned short*>(VtU);
  unsigned short* P16    = reinterpret_cast<unsigned short*>(PldsU);

  const int nqblk = LSEQ / 64;
  const int qblk  = blockIdx.x % nqblk;
  const int h     = (blockIdx.x / nqblk) % NUM_HEADS;
  const int b     = blockIdx.x / (nqblk * NUM_HEADS);
  const int kvh   = h / GROUPS;

  const int tid  = threadIdx.x;
  const int lane = tid & 31;
  const int wid  = tid >> 5;
  const int n16  = lane & 15;
  const int hh   = lane >> 4;

  const int q0    = qblk * 64;
  const int qbase = q0 + wid * 16;

  // ---- Q fragments with RoPE, A-matrix layout (two 16x32 f16 chunks: d 0..31 / 32..63)
  const int    qrow = qbase + n16;
  const float  posq = (float)pos[b * LSEQ + qrow];
  const size_t qoff = ((size_t)(b * LSEQ + qrow) * NUM_HEADS + h) * HDIM;

  FragU qa0, qa1;
  #pragma unroll
  for (int ep = 0; ep < 8; ++ep) {
    unsigned int lo0 = 0, hi0 = 0, lo1 = 0, hi1 = 0;
    #pragma unroll
    for (int sub = 0; sub < 2; ++sub) {
      int e  = 2 * ep + sub;
      int d0 = ((e < 8) ? e : e + 8) + 8 * hh;      // 0..31
      float x1 = Q[qoff + d0];
      float x2 = Q[qoff + d0 + 32];
      float ts = __expf(-LN1E4_OVER_32 * (float)d0); // 10000^(-d0/32)
      float sn, cs;
      __sincosf(posq * ts, &sn, &cs);
      float r1 = (x1 * cs - x2 * sn) * 0.125f;       // fold 1/sqrt(64)
      float r2 = (x2 * cs + x1 * sn) * 0.125f;
      if (sub == 0) { lo0 = f2h(r1); lo1 = f2h(r2); }
      else          { hi0 = f2h(r1); hi1 = f2h(r2); }
    }
    qa0.u[ep] = lo0 | (hi0 << 16);
    qa1.u[ep] = lo1 | (hi1 << 16);
  }

  // all-ones B fragment: 5th PV gemm accumulates softmax row-sums into o4
  FragU ones;
  #pragma unroll
  for (int r = 0; r < 8; ++r) ones.u[r] = 0x3C003C00u;  // (1.0h, 1.0h)

  // ---- accumulators & softmax state (C layout: VGPR r <-> M=r+8*hh, N=n16)
  v8f o0, o1, o2, o3, o4;
  #pragma unroll
  for (int i = 0; i < 8; ++i) { o0[i]=0.f; o1[i]=0.f; o2[i]=0.f; o3[i]=0.f; o4[i]=0.f; }
  float mrow[8];
  #pragma unroll
  for (int r = 0; r < 8; ++r) mrow[r] = NEG_BIG;

  // loop-invariant fill indexing: 128 == 0 (mod 32/64)
  const float kts = __expf(-LN1E4_OVER_32 * (float)lane);  // K-fill timescale (d0 = lane)
  const int   vd  = tid & 63;                              // V-fill dim
  const int   vkp = tid >> 6;                              // V-fill key parity

  const int nkt = qblk * 2 + 2;          // key tiles of 32, covering keys 0 .. q0+63
  for (int kt = 0; kt < nkt; ++kt) {
    const int k0 = kt * 32;
    __syncthreads();                     // previous iteration's LDS reads complete

    // ---- cooperative LDS fill: RoPE'd K (f16) and V^T (f16)
    #pragma unroll
    for (int i = 0; i < 8; ++i) {
      int key = 4 * i + wid;             // 32 keys, d0 = lane
      int kk  = k0 + key;
      size_t koff = ((size_t)(b * LSEQ + kk) * NUM_KV + kvh) * HDIM;
      float x1 = K[koff + lane];
      float x2 = K[koff + lane + 32];
      float sn, cs;
      __sincosf((float)pos[b * LSEQ + kk] * kts, &sn, &cs);
      Klds16[key * KSTR_H + lane]      = f2h(x1 * cs - x2 * sn);
      Klds16[key * KSTR_H + lane + 32] = f2h(x2 * cs + x1 * sn);
    }
    #pragma unroll
    for (int i = 0; i < 16; ++i) {
      int key = 2 * i + vkp;             // 32 keys, d = vd
      int kk  = k0 + key;
      float x = V[((size_t)(b * LSEQ + kk) * NUM_KV + kvh) * HDIM + vd];
      Vt16[vd * VSTR_H + key] = f2h(x);
    }
    __syncthreads();

    // ---- prefetch next tile's K/V rows (overlaps with compute below)
    if (kt + 1 < nkt) {
      int kk = (kt + 1) * 32 + (tid & 31);
      const char* kp = (const char*)&K[((size_t)(b * LSEQ + kk) * NUM_KV + kvh) * HDIM];
      const char* vp = (const char*)&V[((size_t)(b * LSEQ + kk) * NUM_KV + kvh) * HDIM];
      const char* p  = (wid & 2) ? vp : kp;
      __builtin_prefetch(p + (wid & 1) * 128, 0, 3);
    }

    // ---- S = Q * K^T : two 16x16 key subtiles, K-dim = 64 (2 chunks of 32)
    v8f s0, s1;
    #pragma unroll
    for (int i = 0; i < 8; ++i) { s0[i] = 0.f; s1[i] = 0.f; }
    #pragma unroll
    for (int sub = 0; sub < 2; ++sub) {
      int keyl = sub * 16 + n16;         // B layout: N = key, dword r <-> d pair 2*((r<4)?r:r+4)+8*hh
      FragU kb0, kb1;
      #pragma unroll
      for (int r = 0; r < 8; ++r) {
        kb0.u[r] = KldsU[keyl * KSTR_DW + ((r < 4) ? r : r + 4) + 4 * hh];        // d 0..31
        kb1.u[r] = KldsU[keyl * KSTR_DW + 16 + ((r < 4) ? r : r + 4) + 4 * hh];   // d 32..63
      }
      v8f acc = (sub == 0) ? s0 : s1;
      acc = __builtin_amdgcn_wmma_f32_16x16x32_f16(false, qa0.h, false, kb0.h,
                                                   (short)0, acc, false, false);
      acc = __builtin_amdgcn_wmma_f32_16x16x32_f16(false, qa1.h, false, kb1.h,
                                                   (short)0, acc, false, false);
      if (sub == 0) s0 = acc; else s1 = acc;
    }

    // ---- causal mask + online softmax (row max via DPP butterflies)
    const int key0 = k0 + n16;
    const int key1 = key0 + 16;
    #pragma unroll
    for (int r = 0; r < 8; ++r) {
      int q = qbase + r + 8 * hh;
      float v0 = (key0 <= q) ? s0[r] : NEG_BIG;
      float v1 = (key1 <= q) ? s1[r] : NEG_BIG;
      float mx = row16_max(fmaxf(v0, v1));
      float mn   = fmaxf(mrow[r], mx);
      float corr = __expf(mrow[r] - mn);
      mrow[r] = mn;
      float p0 = __expf(v0 - mn);
      float p1 = __expf(v1 - mn);
      o0[r] *= corr; o1[r] *= corr; o2[r] *= corr; o3[r] *= corr; o4[r] *= corr;
      int M = r + 8 * hh;                // stage P (C layout -> LDS row-major)
      P16[(wid * 16 + M) * PSTR_H + n16]      = f2h(p0);
      P16[(wid * 16 + M) * PSTR_H + 16 + n16] = f2h(p1);
    }

    // ---- reload P as A fragment (16x32, K-dim = key slot)
    FragU pa;
    #pragma unroll
    for (int r = 0; r < 8; ++r) {
      int kb = ((r < 4) ? r : r + 4) + 4 * hh;     // dword of key pair per A layout
      pa.u[r] = PldsU[(wid * 16 + n16) * PSTR_DW + kb];
    }

    // ---- O += P * V  (4 N-chunks of 16 over D=64) + row-sum via ones column
    #pragma unroll
    for (int n = 0; n < 4; ++n) {
      FragU vb;                                    // B layout: N = d, K = key slot
      #pragma unroll
      for (int r = 0; r < 8; ++r)
        vb.u[r] = VtU[(n * 16 + n16) * VSTR_DW + ((r < 4) ? r : r + 4) + 4 * hh];
      v8f acc = (n == 0) ? o0 : (n == 1) ? o1 : (n == 2) ? o2 : o3;
      acc = __builtin_amdgcn_wmma_f32_16x16x32_f16(false, pa.h, false, vb.h,
                                                   (short)0, acc, false, false);
      if (n == 0) o0 = acc; else if (n == 1) o1 = acc;
      else if (n == 2) o2 = acc; else o3 = acc;
    }
    o4 = __builtin_amdgcn_wmma_f32_16x16x32_f16(false, pa.h, false, ones.h,
                                                (short)0, o4, false, false);
  }

  // ---- normalize rows and store fp32 output [B, L, H*D]
  #pragma unroll
  for (int r = 0; r < 8; ++r) {
    int q = qbase + r + 8 * hh;
    float inv = 1.0f / o4[r];            // every lane holds its rows' denominator
    size_t obase = (size_t)(b * LSEQ + q) * (NUM_HEADS * HDIM) + h * HDIM + n16;
    out[obase +  0] = o0[r] * inv;
    out[obase + 16] = o1[r] * inv;
    out[obase + 32] = o2[r] * inv;
    out[obase + 48] = o3[r] * inv;
  }
}

extern "C" void kernel_launch(void* const* d_in, const int* in_sizes, int n_in,
                              void* d_out, int out_size, void* d_ws, size_t ws_size,
                              hipStream_t stream) {
  const float* Q   = (const float*)d_in[0];
  const float* K   = (const float*)d_in[1];
  const float* V   = (const float*)d_in[2];
  const int*   pos = (const int*)d_in[3];
  // d_in[4] (attention_mask) is implicit causal-by-index; not needed.
  float* out = (float*)d_out;

  int B = in_sizes[3] / LSEQ;                 // positions is [B, L]
  dim3 grid(B * NUM_HEADS * (LSEQ / 64));
  dim3 block(128);                            // 4 waves (wave32) per WG
  fa_rope_wmma_kernel<<<grid, block, 0, stream>>>(Q, K, V, pos, out);
}
